// CrossAttention_7584912244947
// MI455X (gfx1250) — matrix operand order repaired and checked by reading, not compiled, not run on hardware
//
#include <hip/hip_runtime.h>
#include <hip/hip_bf16.h>
#include <cstdint>

// ---------------------------------------------------------------------------
// CDNA5 (gfx1250) causal self-attention with RoPE on v_wmma_f32_16x16x32_bf16.
// All LDS tiles are staged as contiguous b128 copies (weights pre-transposed),
// through GLOBAL_LOAD_ASYNC_TO_LDS_B128 (ASYNCcnt) when available.
// ---------------------------------------------------------------------------

typedef __attribute__((ext_vector_type(16))) __bf16 v16bf;
typedef __attribute__((ext_vector_type(8)))  float  v8f;
typedef __attribute__((ext_vector_type(4)))  int    v4i;

struct __align__(16) U4 { unsigned int a[4]; };
struct __align__(16) F4 { float f[4]; };

union Frag { v16bf v; U4 q[2]; };

__device__ __forceinline__ v8f wmma_bf16(v16bf a, v16bf b, v8f c) {
  // 8 args: (neg_a, A, neg_b, B, c_mod, C, reuse_a, reuse_b)
  return __builtin_amdgcn_wmma_f32_16x16x32_bf16(false, a, false, b,
                                                 (short)0, c, false, false);
}

// ---- async global->LDS staging (gfx1250 ASYNCcnt path), guarded ----------
#define ASG __attribute__((address_space(1)))
#define ASL __attribute__((address_space(3)))

#if __has_builtin(__builtin_amdgcn_global_load_async_to_lds_b128)
#define HAVE_ASYNC 1
// copy 16 bytes per lane: lane's global addr -> lane's LDS addr
// signature (probe-confirmed): (v4i AS1*, v4i AS3*, imm offset, imm cpol)
#define CP_B128(ldsp, gp)                                                  \
  __builtin_amdgcn_global_load_async_to_lds_b128(                          \
      (ASG v4i*)(uintptr_t)(gp),                                           \
      (ASL v4i*)(unsigned)(uintptr_t)(ldsp), 0, 0)
#else
#define HAVE_ASYNC 0
#endif

__device__ __forceinline__ void wait_async() {
#if HAVE_ASYNC
#if __has_builtin(__builtin_amdgcn_s_wait_asynccnt)
  __builtin_amdgcn_s_wait_asynccnt(0);
#else
  asm volatile("s_wait_asynccnt 0x0" ::: "memory");
#endif
#endif
}

// ---------------------------------------------------------------------------
// f32 -> bf16 bulk convert (b128 load, b64 store)
// ---------------------------------------------------------------------------
__global__ void cvt_f32_to_bf16(const float* __restrict__ s,
                                __bf16* __restrict__ d, int n4)
{
  int i = blockIdx.x * blockDim.x + threadIdx.x;
  if (i >= n4) return;
  F4 f = ((const F4*)s)[i];
  union { __bf16 e[4]; unsigned long long u; } o;
  o.e[0] = (__bf16)f.f[0]; o.e[1] = (__bf16)f.f[1];
  o.e[2] = (__bf16)f.f[2]; o.e[3] = (__bf16)f.f[3];
  ((unsigned long long*)d)[i] = o.u;
}

// f32 (K,N) -> bf16 transposed (N,K); one-time 1M-element weight prep so the
// GEMM B tiles become row-contiguous (no software transpose in the hot loop).
__global__ void cvt_w_transpose(const float* __restrict__ s,
                                __bf16* __restrict__ d)
{
  int idx = blockIdx.x * blockDim.x + threadIdx.x;  // over 1024*1024
  int k = idx >> 10, n = idx & 1023;
  d[((size_t)n << 10) + k] = (__bf16)s[idx];
}

// ---------------------------------------------------------------------------
// 128x128 tiled bf16 GEMM, K-step 32, double-buffered LDS, 8 waves (4x2),
// each wave 32x64 = 8 WMMA tiles per k-step.
//   A  : (8192 x 1024) row-major bf16
//   BT : (1024 x 1024) bf16, TRANSPOSED weights (N,K) -> contiguous staging
// MODE 1: RoPE + head-split epilogue (Q/K -> (bh,T,64))
// MODE 2: head-split TRANSPOSED epilogue (V -> (bh,64,T))
// MODE 3: f32 + bias epilogue to d_out
// ---------------------------------------------------------------------------
template<int MODE>
__global__ __launch_bounds__(256) void gemm128(const __bf16* __restrict__ A,
                                               const __bf16* __restrict__ BT,
                                               __bf16* __restrict__ outB,
                                               float* __restrict__ outF,
                                               const float* __restrict__ bias)
{
  constexpr int K = 1024, N = 1024;
  __shared__ __align__(16) __bf16 As[2][128][40];
  __shared__ __align__(16) __bf16 Bs[2][128][40];   // [n][k], already transposed

  const int tid  = threadIdx.x;
  const int wid  = tid >> 5, lane = tid & 31;
  const int m0   = blockIdx.y * 128, n0 = blockIdx.x * 128;
  const int wm   = (wid & 3) * 32,  wn = (wid >> 2) * 64;
  const int rsel = lane & 15;
  const int half8 = (lane < 16) ? 0 : 8;    // A-frag K sub-offset per lane half
  const int kb16  = (lane < 16) ? 0 : 16;   // unused; B-frag uses half8 like A

  v8f acc[2][4];
#pragma unroll
  for (int i = 0; i < 2; ++i)
#pragma unroll
    for (int j = 0; j < 4; ++j)
#pragma unroll
      for (int r = 0; r < 8; ++r) acc[i][j][r] = 0.0f;

  // staging map: 256 threads, each copies 16 elems of one row (A and B alike)
  const int lrow = tid >> 1, lseg = (tid & 1) * 16;
  const __bf16* Ag = A  + (size_t)(m0 + lrow) * K + lseg;
  const __bf16* Bg = BT + (size_t)(n0 + lrow) * K + lseg;

  auto stage = [&](int buf, int k0) {
#if HAVE_ASYNC
    CP_B128(&As[buf][lrow][lseg],     Ag + k0);
    CP_B128(&As[buf][lrow][lseg + 8], Ag + k0 + 8);
    CP_B128(&Bs[buf][lrow][lseg],     Bg + k0);
    CP_B128(&Bs[buf][lrow][lseg + 8], Bg + k0 + 8);
#else
    const U4* ga = (const U4*)(Ag + k0);
    U4 a0 = ga[0], a1 = ga[1];
    *(U4*)&As[buf][lrow][lseg]     = a0;
    *(U4*)&As[buf][lrow][lseg + 8] = a1;
    const U4* gb = (const U4*)(Bg + k0);
    U4 b0 = gb[0], b1 = gb[1];
    *(U4*)&Bs[buf][lrow][lseg]     = b0;
    *(U4*)&Bs[buf][lrow][lseg + 8] = b1;
#endif
  };

  stage(0, 0);
  wait_async();
  __syncthreads();

  int cur = 0;
  for (int k0 = 0; k0 < K; k0 += 32) {
    if (k0 + 32 < K) stage(cur ^ 1, k0 + 32);       // overlap next tile
    if (k0 + 64 < K) {                               // cover HBM latency
      __builtin_prefetch(Ag + k0 + 64, 0, 1);
      __builtin_prefetch(Bg + k0 + 64, 0, 1);
    }

    Frag af[2];
#pragma unroll
    for (int mi = 0; mi < 2; ++mi) {
      const int row = wm + mi * 16 + rsel;
      af[mi].q[0] = *(const U4*)&As[cur][row][half8];
      af[mi].q[1] = *(const U4*)&As[cur][row][16 + half8];
    }
#pragma unroll
    for (int ni = 0; ni < 4; ++ni) {
      Frag bf;
      const int col = wn + ni * 16 + rsel;
      bf.q[0] = *(const U4*)&Bs[cur][col][half8];
      bf.q[1] = *(const U4*)&Bs[cur][col][16 + half8];
#pragma unroll
      for (int mi = 0; mi < 2; ++mi)
        acc[mi][ni] = wmma_bf16(af[mi].v, bf.v, acc[mi][ni]);
    }

    wait_async();
    __syncthreads();
    cur ^= 1;
  }

  // ---- epilogue --------------------------------------------------------
#pragma unroll
  for (int mi = 0; mi < 2; ++mi)
#pragma unroll
    for (int ni = 0; ni < 4; ++ni)
#pragma unroll
      for (int r = 0; r < 8; ++r) {
        const int grow = m0 + wm + mi * 16 + r + half8;   // global M row
        const int gcol = n0 + wn + ni * 16 + rsel;        // global N col
        float v = acc[mi][ni][r];
        if (MODE == 3) {
          outF[(size_t)grow * N + gcol] = v + bias[gcol];
        } else if (MODE == 2) {
          // V: (bh, d, T) so flash can stage V tiles contiguously
          const int b = grow >> 11, t = grow & 2047;
          const int h = gcol >> 6,  dH = gcol & 63;
          outB[((size_t)(b * 16 + h) * 64 + dH) * 2048 + t] = (__bf16)v;
        } else {  // MODE == 1 : RoPE (even/odd interleave) + head split
          float partner = __shfl_xor(v, 1, 32);  // pair lives in lane^1
          const int b = grow >> 11, t = grow & 2047;
          const int h = gcol >> 6,  dH = gcol & 63;
          const int pi = dH >> 1;
          float invf = __expf(-(float)(2 * pi) * (9.210340371976184f / 64.0f));
          float sn, cs;
          __sincosf((float)t * invf, &sn, &cs);
          float x1 = (dH & 1) ? partner : v;
          float x2 = (dH & 1) ? v : partner;
          float rv = (dH & 1) ? (x1 * sn + x2 * cs) : (x1 * cs - x2 * sn);
          outB[(((size_t)b * 16 + h) * 2048 + t) * 64 + dH] = (__bf16)rv;
        }
      }
}

// ---------------------------------------------------------------------------
// Flash attention, causal, d=64.  128 threads (4 waves), 64 q-rows per block,
// 32-key chunks in LDS, online softmax, WMMA for S=QK^T and O+=P*V.
// Q,K: (bh, T, 64) bf16;  V: (bh, 64, T) bf16 (pre-transposed by gemm MODE 2).
// ---------------------------------------------------------------------------
__global__ __launch_bounds__(128) void flash64(const __bf16* __restrict__ Qh,
                                               const __bf16* __restrict__ Kh,
                                               const __bf16* __restrict__ Vh,
                                               __bf16* __restrict__ Ab)
{
  constexpr int T = 2048, DH = 64;
  __shared__ __align__(16) __bf16 Qs[64][72];
  __shared__ __align__(16) __bf16 Ks[32][72];
  __shared__ __align__(16) __bf16 VsT[64][40];    // [d][key]
  __shared__ __align__(16) __bf16 Ps[4][16][40];  // per-wave P relayout scratch

  const int bh    = blockIdx.y;
  const int qbase = blockIdx.x * 64;
  const int tid = threadIdx.x, wid = tid >> 5, lane = tid & 31;
  const int rsel  = lane & 15;
  const int half8 = (lane < 16) ? 0 : 8;
  const int kb16  = (lane < 16) ? 0 : 16;

  const __bf16* Qp = Qh + ((size_t)bh * T + qbase) * DH;
  const __bf16* Kp = Kh + (size_t)bh * T * DH;
  const __bf16* Vp = Vh + (size_t)bh * DH * T;

  // cooperative Q tile: 64 x 64
  {
    const int r = tid >> 1, s = (tid & 1) * 32;
#if HAVE_ASYNC
#pragma unroll
    for (int j = 0; j < 4; ++j)
      CP_B128(&Qs[r][s + j * 8], Qp + (size_t)r * DH + s + j * 8);
#else
    const U4* g = (const U4*)(Qp + (size_t)r * DH + s);
#pragma unroll
    for (int j = 0; j < 4; ++j) *(U4*)&Qs[r][s + j * 8] = g[j];
#endif
  }
  wait_async();
  __syncthreads();

  Frag qf[2];
#pragma unroll
  for (int dc = 0; dc < 2; ++dc) {
    const int row = wid * 16 + rsel;
    qf[dc].q[0] = *(const U4*)&Qs[row][dc * 32 + half8];
    qf[dc].q[1] = *(const U4*)&Qs[row][dc * 32 + 16 + half8];
  }

  v8f o[4];
  float mrow[8], lrow[8];
#pragma unroll
  for (int dt = 0; dt < 4; ++dt)
#pragma unroll
    for (int r = 0; r < 8; ++r) o[dt][r] = 0.0f;
#pragma unroll
  for (int r = 0; r < 8; ++r) { mrow[r] = -1.0e30f; lrow[r] = 0.0f; }

  const int kr = tid >> 2, ks2 = (tid & 3) * 16;   // K stage map (32x64)
  const int vd = tid >> 1, vs = (tid & 1) * 16;    // V stage map (64x32)

  const int nchunk = qbase / 32 + 2;  // causal: keys 0 .. qbase+63
  for (int ch = 0; ch < nchunk; ++ch) {
    const int kstart = ch * 32;
    __syncthreads();  // previous chunk fully consumed before overwrite
#if HAVE_ASYNC
    CP_B128(&Ks[kr][ks2],      Kp + (size_t)(kstart + kr) * DH + ks2);
    CP_B128(&Ks[kr][ks2 + 8],  Kp + (size_t)(kstart + kr) * DH + ks2 + 8);
    CP_B128(&VsT[vd][vs],      Vp + (size_t)vd * T + kstart + vs);
    CP_B128(&VsT[vd][vs + 8],  Vp + (size_t)vd * T + kstart + vs + 8);
#else
    {
      const U4* g = (const U4*)(Kp + (size_t)(kstart + kr) * DH + ks2);
      U4 t0 = g[0], t1 = g[1];
      *(U4*)&Ks[kr][ks2]     = t0;
      *(U4*)&Ks[kr][ks2 + 8] = t1;
    }
    {
      const U4* g = (const U4*)(Vp + (size_t)vd * T + kstart + vs);
      U4 t0 = g[0], t1 = g[1];
      *(U4*)&VsT[vd][vs]     = t0;
      *(U4*)&VsT[vd][vs + 8] = t1;
    }
#endif
    wait_async();
    __syncthreads();

    // S = Q * K^T : two 16x16 key tiles, reduce over d = 64 (2 WMMA each)
    v8f s0, s1;
#pragma unroll
    for (int r = 0; r < 8; ++r) { s0[r] = 0.0f; s1[r] = 0.0f; }
#pragma unroll
    for (int dc = 0; dc < 2; ++dc) {
      Frag kf0, kf1;
      kf0.q[0] = *(const U4*)&Ks[rsel][dc * 32 + half8];
      kf0.q[1] = *(const U4*)&Ks[rsel][dc * 32 + 16 + half8];
      kf1.q[0] = *(const U4*)&Ks[16 + rsel][dc * 32 + half8];
      kf1.q[1] = *(const U4*)&Ks[16 + rsel][dc * 32 + 16 + half8];
      s0 = wmma_bf16(qf[dc].v, kf0.v, s0);
      s1 = wmma_bf16(qf[dc].v, kf1.v, s1);
    }

    // online softmax (row r+half8; 16 lanes of one half share a row)
    constexpr float scale = 0.125f;  // 1/sqrt(64)
#pragma unroll
    for (int r = 0; r < 8; ++r) {
      const int qrow = qbase + wid * 16 + r + half8;
      const int key0 = kstart + rsel, key1 = kstart + 16 + rsel;
      float v0 = (key0 <= qrow) ? s0[r] * scale : -1.0e30f;
      float v1 = (key1 <= qrow) ? s1[r] * scale : -1.0e30f;
      float mx = fmaxf(v0, v1);
#pragma unroll
      for (int off = 1; off < 16; off <<= 1)
        mx = fmaxf(mx, __shfl_xor(mx, off, 32));
      const float mn  = fmaxf(mrow[r], mx);
      const float fac = __expf(mrow[r] - mn);
      const float p0  = __expf(v0 - mn);
      const float p1  = __expf(v1 - mn);
      float sum = p0 + p1;
#pragma unroll
      for (int off = 1; off < 16; off <<= 1)
        sum += __shfl_xor(sum, off, 32);
      lrow[r] = lrow[r] * fac + sum;
      mrow[r] = mn;
#pragma unroll
      for (int dt = 0; dt < 4; ++dt) o[dt][r] *= fac;
      // C-layout -> LDS so P can be re-read in the A-fragment layout
      Ps[wid][r + half8][rsel]      = (__bf16)p0;
      Ps[wid][r + half8][16 + rsel] = (__bf16)p1;
    }

    // O += P(16x32) * V(32x64)
    Frag pf;
    pf.q[0] = *(const U4*)&Ps[wid][rsel][half8];
    pf.q[1] = *(const U4*)&Ps[wid][rsel][16 + half8];
#pragma unroll
    for (int dt = 0; dt < 4; ++dt) {
      Frag vf;
      const int dcol = dt * 16 + rsel;
      vf.q[0] = *(const U4*)&VsT[dcol][half8];
      vf.q[1] = *(const U4*)&VsT[dcol][16 + half8];
      o[dt] = wmma_bf16(pf.v, vf.v, o[dt]);
    }
  }

  // normalize and merge heads back to (B, T, H*64)
  const int b = bh >> 4, h = bh & 15;
#pragma unroll
  for (int dt = 0; dt < 4; ++dt)
#pragma unroll
    for (int r = 0; r < 8; ++r) {
      const int qrow = qbase + wid * 16 + r + half8;
      const float outv = o[dt][r] / lrow[r];
      Ab[((size_t)b * T + qrow) * 1024 + h * 64 + dt * 16 + rsel] = (__bf16)outv;
    }
}

// ---------------------------------------------------------------------------
// Host-side orchestration (graph-capture safe: launches only)
// ---------------------------------------------------------------------------
extern "C" void kernel_launch(void* const* d_in, const int* in_sizes, int n_in,
                              void* d_out, int out_size, void* d_ws, size_t ws_size,
                              hipStream_t stream)
{
  (void)in_sizes; (void)n_in; (void)out_size; (void)ws_size;

  const float* x  = (const float*)d_in[0];
  const float* Wq = (const float*)d_in[1];
  const float* Wk = (const float*)d_in[2];
  const float* Wv = (const float*)d_in[3];
  const float* Wo = (const float*)d_in[4];
  const float* bo = (const float*)d_in[5];
  float* out = (float*)d_out;

  char* ws = (char*)d_ws;
  const size_t MB = 1024 * 1024;
  __bf16* xb  = (__bf16*)(ws);            // 16 MB  x bf16 (8192x1024)
  __bf16* Wqb = (__bf16*)(ws + 16 * MB);  //  2 MB  transposed (N,K)
  __bf16* Wkb = (__bf16*)(ws + 18 * MB);  //  2 MB  transposed
  __bf16* Wvb = (__bf16*)(ws + 20 * MB);  //  2 MB  transposed
  __bf16* Wob = (__bf16*)(ws + 22 * MB);  //  2 MB  transposed
  __bf16* Qh  = (__bf16*)(ws + 24 * MB);  // 16 MB  (bh, T, 64) post-RoPE
  __bf16* Kh  = (__bf16*)(ws + 40 * MB);  // 16 MB  (bh, T, 64) post-RoPE
  __bf16* Vh  = (__bf16*)(ws + 56 * MB);  // 16 MB  (bh, 64, T)
  __bf16* Ab  = (__bf16*)(ws + 72 * MB);  // 16 MB  attn out (B,T,1024)

  // conversions (weights transposed once so GEMM staging is contiguous)
  cvt_f32_to_bf16<<<8192, 256, 0, stream>>>(x, xb, 2097152);
  cvt_w_transpose<<<4096, 256, 0, stream>>>(Wq, Wqb);
  cvt_w_transpose<<<4096, 256, 0, stream>>>(Wk, Wkb);
  cvt_w_transpose<<<4096, 256, 0, stream>>>(Wv, Wvb);
  cvt_w_transpose<<<4096, 256, 0, stream>>>(Wo, Wob);

  // projections (RoPE fused for Q/K; V stored head-transposed)
  dim3 gg(1024 / 128, 8192 / 128);
  gemm128<1><<<gg, 256, 0, stream>>>(xb, Wqb, Qh, nullptr, nullptr);
  gemm128<1><<<gg, 256, 0, stream>>>(xb, Wkb, Kh, nullptr, nullptr);
  gemm128<2><<<gg, 256, 0, stream>>>(xb, Wvb, Vh, nullptr, nullptr);

  // causal flash attention: 32 q-blocks x 64 (b,h)
  flash64<<<dim3(32, 64), 128, 0, stream>>>(Qh, Kh, Vh, Ab);

  // output projection + bias -> f32 d_out
  gemm128<3><<<gg, 256, 0, stream>>>(Ab, Wob, nullptr, out, bo);
}
